// RelativePosAttention_42992622633323
// MI455X (gfx1250) — matrix operand order repaired and checked by reading, not compiled, not run on hardware
//
#include <hip/hip_runtime.h>
#include <hip/hip_bf16.h>
#include <stdint.h>

// ---- problem constants ----
#define PB 2
#define PN 2048
#define PC 512
#define PH 8
#define PHD 64

typedef __attribute__((ext_vector_type(16))) __bf16 v16bf;
typedef __attribute__((ext_vector_type(8)))  float  v8f;

union FragU { int i[8]; v16bf v; };
union H2U { _Float16 h; uint16_t u; };

__device__ __forceinline__ uint16_t f2bf_rne(float f) {
  uint32_t u = __builtin_bit_cast(uint32_t, f);
  uint32_t r = u + 0x7FFFu + ((u >> 16) & 1u);
  return (uint16_t)(r >> 16);
}

__device__ __forceinline__ uint16_t f2h(float f) {
  H2U c; c.h = (_Float16)f; return c.u;
}
__device__ __forceinline__ float h2f(uint16_t u) {
  H2U c; c.u = u; return (float)c.h;
}

// A fragment 16x32 bf16: p -> element [row0, k0] of row-major [M,K], ld = K stride
__device__ __forceinline__ v16bf load_a_frag(const uint16_t* p, int ld) {
  int lane = threadIdx.x & 31;
  int g = lane >> 4, m = lane & 15;
  const uint16_t* row = p + (size_t)m * ld;
  FragU f;
#pragma unroll
  for (int r = 0; r < 8; ++r) {
    int k = ((r >> 2) << 4) + (g << 3) + ((r & 3) << 1);
    f.i[r] = *(const int*)(row + k);
  }
  return f.v;
}

// B fragment 32x16 bf16 from B^T stored row-major [N,K]: p -> element [n0, k0]
__device__ __forceinline__ v16bf load_b_frag(const uint16_t* p, int ld) {
  int lane = threadIdx.x & 31;
  int g = lane >> 4, n = lane & 15;
  const uint16_t* row = p + (size_t)n * ld;
  FragU f;
#pragma unroll
  for (int r = 0; r < 8; ++r) {
    int k = ((r >> 2) << 4) + (g << 3) + ((r & 3) << 1);
    f.i[r] = *(const int*)(row + k);
  }
  return f.v;
}

__device__ __forceinline__ v8f wmma_bf16(v16bf a, v16bf b, v8f c) {
  return __builtin_amdgcn_wmma_f32_16x16x32_bf16(false, a, false, b, (short)0, c,
                                                 false, false);
}

// ---------------- fp32 -> bf16 convert ----------------
__global__ void cvt_f32_bf16(const float* __restrict__ s, uint16_t* __restrict__ d, int n) {
  int i = blockIdx.x * blockDim.x + threadIdx.x;
  if (i < n) d[i] = f2bf_rne(s[i]);
}

// ---------------- generic projection GEMM: out = A[M,K] @ W[Nout,K]^T ----------------
// mode 0: QKV epilogue (writes qu,qv,k,vT per-head); mode 1: pos epilogue; mode 2: fp32 out + bias
__global__ __launch_bounds__(128) void gemm_proj(
    const uint16_t* __restrict__ A, const uint16_t* __restrict__ Wt,
    int M, int K, int Nout, int mode,
    const float* __restrict__ ub, const float* __restrict__ vb,
    const float* __restrict__ bproj,
    uint16_t* __restrict__ qu, uint16_t* __restrict__ qv,
    uint16_t* __restrict__ kb, uint16_t* __restrict__ vT,
    uint16_t* __restrict__ posb, float* __restrict__ outf) {
  int wave = blockIdx.x * (blockDim.x >> 5) + (threadIdx.x >> 5);
  int ntiles = Nout >> 6;
  int mt = wave / ntiles;
  int nq = wave % ntiles;
  int lane = threadIdx.x & 31;
  v8f acc[4] = {};
  const uint16_t* Arow = A + (size_t)mt * 16 * K;
  const uint16_t* Wrow = Wt + (size_t)(nq * 64) * K;
  for (int k0 = 0; k0 < K; k0 += 32) {
    // prefetch next k-chunk (speculative, safe)
    __builtin_prefetch(Arow + (size_t)(lane & 15) * K + k0 + 32, 0, 3);
    __builtin_prefetch(Wrow + (size_t)lane * K + k0 + 32, 0, 3);
    __builtin_prefetch(Wrow + (size_t)(lane + 32) * K + k0 + 32, 0, 3);
    // all fragment loads hoisted ahead of the WMMAs (partial waits, overlap)
    v16bf a  = load_a_frag(Arow + k0, K);
    v16bf b0 = load_b_frag(Wrow + (size_t)0  * K + k0, K);
    v16bf b1 = load_b_frag(Wrow + (size_t)16 * K + k0, K);
    v16bf b2 = load_b_frag(Wrow + (size_t)32 * K + k0, K);
    v16bf b3 = load_b_frag(Wrow + (size_t)48 * K + k0, K);
    acc[0] = wmma_bf16(a, b0, acc[0]);
    acc[1] = wmma_bf16(a, b1, acc[1]);
    acc[2] = wmma_bf16(a, b2, acc[2]);
    acc[3] = wmma_bf16(a, b3, acc[3]);
  }
  int g = lane >> 4, n = lane & 15;
#pragma unroll
  for (int t = 0; t < 4; ++t) {
#pragma unroll
    for (int r = 0; r < 8; ++r) {
      float vacc = acc[t][r];
      int row = mt * 16 + r + 8 * g;
      int col = nq * 64 + t * 16 + n;
      if (mode == 0) {
        int sel = col / PC;
        int cc = col % PC;
        int h = cc / PHD, d = cc % PHD;
        int b_ = row / PN, ni = row % PN;
        size_t hoff = ((size_t)(b_ * PH + h) * PN + ni) * PHD + d;
        if (sel == 0) {
          qu[hoff] = f2bf_rne(vacc + ub[h * PHD + d]);
          qv[hoff] = f2bf_rne(vacc + vb[h * PHD + d]);
        } else if (sel == 1) {
          kb[hoff] = f2bf_rne(vacc);
        } else {
          vT[((size_t)(b_ * PH + h) * PHD + d) * PN + ni] = f2bf_rne(vacc);
        }
      } else if (mode == 1) {
        int h = col / PHD, d = col % PHD;
        int b_ = row / PN, ni = row % PN;
        posb[((size_t)(b_ * PH + h) * PN + ni) * PHD + d] = f2bf_rne(vacc);
      } else {
        outf[(size_t)row * Nout + col] = vacc + bproj[col];
      }
    }
  }
}

// ---------------- pos-score GEMM with fused relative-shift scatter (fp16 out) ----------------
// ps[i,m] = qv[b,h,i,:] . pos[b,h,m,:]  (K = 64)
// shifted[i, m-(N-1)+i] = ps[i,m]   if m >= N-1-i
// shifted[i-1, m+i+1]   = ps[i,m]   if m <= N-2-i and i >= 1   (else dropped)
__global__ __launch_bounds__(128) void gemm_ps_shift(
    const uint16_t* __restrict__ qvb, const uint16_t* __restrict__ posb,
    uint16_t* __restrict__ S) {
  int wave = blockIdx.x * (blockDim.x >> 5) + (threadIdx.x >> 5);
  const int kq = PN >> 6;   // key-column quads of 64
  const int qt = PN >> 4;   // query tiles of 16
  int bh = wave / (kq * qt);
  int rem = wave % (kq * qt);
  int it = rem / kq;
  int mq = rem % kq;
  const uint16_t* Aq = qvb + (size_t)bh * PN * PHD + (size_t)it * 16 * PHD;
  const uint16_t* Bp = posb + (size_t)bh * PN * PHD + (size_t)(mq * 64) * PHD;
  v8f acc[4] = {};
#pragma unroll
  for (int k0 = 0; k0 < PHD; k0 += 32) {
    v16bf a  = load_a_frag(Aq + k0, PHD);
    v16bf b0 = load_b_frag(Bp + (size_t)0  * PHD + k0, PHD);
    v16bf b1 = load_b_frag(Bp + (size_t)16 * PHD + k0, PHD);
    v16bf b2 = load_b_frag(Bp + (size_t)32 * PHD + k0, PHD);
    v16bf b3 = load_b_frag(Bp + (size_t)48 * PHD + k0, PHD);
    acc[0] = wmma_bf16(a, b0, acc[0]);
    acc[1] = wmma_bf16(a, b1, acc[1]);
    acc[2] = wmma_bf16(a, b2, acc[2]);
    acc[3] = wmma_bf16(a, b3, acc[3]);
  }
  int lane = threadIdx.x & 31;
  int g = lane >> 4, n = lane & 15;
  uint16_t* Sbh = S + (size_t)bh * PN * PN;
#pragma unroll
  for (int t = 0; t < 4; ++t) {
#pragma unroll
    for (int r = 0; r < 8; ++r) {
      int i = it * 16 + r + 8 * g;
      int m = mq * 64 + t * 16 + n;
      float vv = acc[t][r];
      if (m >= PN - 1 - i) {
        Sbh[(size_t)i * PN + (m - (PN - 1) + i)] = f2h(vv);
      } else if (i >= 1) {
        Sbh[(size_t)(i - 1) * PN + (m + i + 1)] = f2h(vv);
      }
    }
  }
}

// ---------------- flash attention ----------------
// per wave: 16 query rows; stream keys 32 at a time.
// shifted pos scores (fp16) are double-buffered into LDS with async copies.
__global__ __launch_bounds__(128) void flash_attn(
    const uint16_t* __restrict__ qu, const uint16_t* __restrict__ kb,
    const uint16_t* __restrict__ vT, const uint16_t* __restrict__ S,
    uint16_t* __restrict__ attn_in) {
  __shared__ __align__(16) uint16_t ldsP[4][16 * 32];
  __shared__ __align__(16) uint16_t ldsS[4][2][16 * 32];
  int wslot = threadIdx.x >> 5;
  int wave = blockIdx.x * 4 + wslot;
  const int qt = PN >> 4;
  int bh = wave / qt;
  int it = wave % qt;
  int b_ = bh / PH, h = bh % PH;
  int lane = threadIdx.x & 31;
  int g = lane >> 4, n = lane & 15;

  const uint16_t* Aq = qu + (size_t)bh * PN * PHD + (size_t)it * 16 * PHD;
  v16bf a0 = load_a_frag(Aq + 0, PHD);
  v16bf a1 = load_a_frag(Aq + 32, PHD);

  v8f o[4] = {};
  float mst[8], lst[8];
#pragma unroll
  for (int r = 0; r < 8; ++r) { mst[r] = -1e30f; lst[r] = 0.0f; }

  const uint16_t* Kb = kb + (size_t)bh * PN * PHD;
  const uint16_t* Vt = vT + (size_t)bh * PHD * PN;
  const uint16_t* Sbh = S + (size_t)bh * PN * PN + (size_t)(it * 16) * PN;

  // async copy of one 16x32 fp16 S tile (1 KB) into a per-wave LDS buffer:
  // 2 x (32 lanes x b128) covers 16 rows x 64B.
  auto issue_S = [&](int buf, int j0) {
    uint32_t base = (uint32_t)(uintptr_t)&ldsS[wslot][buf][0];
#pragma unroll
    for (int half = 0; half < 2; ++half) {
      int task = half * 32 + lane;
      int row = task >> 2, seg = task & 3;
      uint32_t loff = base + (uint32_t)(row * 64 + seg * 16);
      uint64_t gaddr = (uint64_t)(uintptr_t)(Sbh + (size_t)row * PN + j0) +
                       (uint64_t)(seg * 16);
      asm volatile("global_load_async_to_lds_b128 %0, %1, off"
                   :: "v"(loff), "v"(gaddr) : "memory");
    }
  };

  issue_S(0, 0);
  for (int j0 = 0; j0 < PN; j0 += 32) {
    int cur = (j0 >> 5) & 1;
    // prefetch next K / V blocks (lanes spread to cover whole blocks)
    __builtin_prefetch(Kb + (size_t)(j0 + 32 + lane) * PHD, 0, 3);
    __builtin_prefetch(Vt + (size_t)lane * PN + j0 + 32, 0, 3);
    __builtin_prefetch(Vt + (size_t)(lane + 32) * PN + j0 + 32, 0, 3);
    // content scores for two 16-key tiles: loads hoisted, then 4 WMMAs
    v16bf bk0 = load_b_frag(Kb + (size_t)j0 * PHD + 0, PHD);
    v16bf bk1 = load_b_frag(Kb + (size_t)j0 * PHD + 32, PHD);
    v16bf bk2 = load_b_frag(Kb + (size_t)(j0 + 16) * PHD + 0, PHD);
    v16bf bk3 = load_b_frag(Kb + (size_t)(j0 + 16) * PHD + 32, PHD);
    v8f s0 = {}, s1 = {};
    s0 = wmma_bf16(a0, bk0, s0);
    s0 = wmma_bf16(a1, bk1, s0);
    s1 = wmma_bf16(a0, bk2, s1);
    s1 = wmma_bf16(a1, bk3, s1);
    // re-issue async S copy for the next block into the other buffer.
    // prior LDS reads of that buffer must be retired first.
    asm volatile("s_wait_dscnt 0x0" ::: "memory");
    issue_S(cur ^ 1, (j0 + 32) & (PN - 1));
    // current buffer's 2 copies are the oldest outstanding -> wait asynccnt<=2
    asm volatile("s_wait_asynccnt 0x2" ::: "memory");
    // add shifted pos scores, scale
    float l0[8], l1[8];
#pragma unroll
    for (int r = 0; r < 8; ++r) {
      int mrow = r + 8 * g;
      int i = it * 16 + mrow;
      float pa = h2f(ldsS[wslot][cur][mrow * 32 + n]);
      float pb = h2f(ldsS[wslot][cur][mrow * 32 + 16 + n]);
      if (j0 + n == i + 1) pa = 0.0f;        // slot never written by shift-scatter
      if (j0 + 16 + n == i + 1) pb = 0.0f;
      l0[r] = (pa + s0[r]) * 0.125f;
      l1[r] = (pb + s1[r]) * 0.125f;
    }
    // online softmax + stage P into per-wave LDS
#pragma unroll
    for (int r = 0; r < 8; ++r) {
      float tm = fmaxf(l0[r], l1[r]);
#pragma unroll
      for (int msk = 1; msk < 16; msk <<= 1) tm = fmaxf(tm, __shfl_xor(tm, msk, 32));
      float nm = fmaxf(mst[r], tm);
      float alpha = __expf(mst[r] - nm);
      mst[r] = nm;
      float p0 = __expf(l0[r] - nm);
      float p1 = __expf(l1[r] - nm);
      float rs = p0 + p1;
#pragma unroll
      for (int msk = 1; msk < 16; msk <<= 1) rs += __shfl_xor(rs, msk, 32);
      lst[r] = lst[r] * alpha + rs;
#pragma unroll
      for (int c = 0; c < 4; ++c) o[c][r] *= alpha;
      int mrow = r + 8 * g;
      ldsP[wslot][mrow * 32 + n] = f2bf_rne(p0);
      ldsP[wslot][mrow * 32 + 16 + n] = f2bf_rne(p1);
    }
    // P as A-fragment from LDS (same-wave LDS is in-order; no barrier needed)
    FragU fa;
    {
      int m = lane & 15;
#pragma unroll
      for (int r = 0; r < 8; ++r) {
        int k = ((r >> 2) << 4) + (g << 3) + ((r & 3) << 1);
        fa.i[r] = *(const int*)&ldsP[wslot][m * 32 + k];
      }
    }
    // O += P @ V  (V stored transposed [HD, N] so B-frags are dword loads)
    v16bf bv0 = load_b_frag(Vt + (size_t)0  * PN + j0, PN);
    v16bf bv1 = load_b_frag(Vt + (size_t)16 * PN + j0, PN);
    v16bf bv2 = load_b_frag(Vt + (size_t)32 * PN + j0, PN);
    v16bf bv3 = load_b_frag(Vt + (size_t)48 * PN + j0, PN);
    o[0] = wmma_bf16(fa.v, bv0, o[0]);
    o[1] = wmma_bf16(fa.v, bv1, o[1]);
    o[2] = wmma_bf16(fa.v, bv2, o[2]);
    o[3] = wmma_bf16(fa.v, bv3, o[3]);
  }
  // normalize and write to [B, N, C] bf16 for the output projection
#pragma unroll
  for (int c = 0; c < 4; ++c) {
#pragma unroll
    for (int r = 0; r < 8; ++r) {
      int i = it * 16 + r + 8 * g;
      int col = h * PHD + c * 16 + n;
      float val = o[c][r] / lst[r];
      attn_in[((size_t)(b_ * PN) + i) * PC + col] = f2bf_rne(val);
    }
  }
}

extern "C" void kernel_launch(void* const* d_in, const int* in_sizes, int n_in,
                              void* d_out, int out_size, void* d_ws, size_t ws_size,
                              hipStream_t stream) {
  (void)in_sizes; (void)n_in; (void)out_size; (void)ws_size;
  const float* x     = (const float*)d_in[0];
  const float* pe    = (const float*)d_in[1];
  const float* Wqkv  = (const float*)d_in[2];
  const float* Wpos  = (const float*)d_in[3];
  const float* ub    = (const float*)d_in[4];
  const float* vb    = (const float*)d_in[5];
  const float* Wproj = (const float*)d_in[6];
  const float* bproj = (const float*)d_in[7];
  float* out = (float*)d_out;

  char* ws = (char*)d_ws;
  size_t off = 0;
  auto alloc = [&](size_t bytes) -> void* {
    void* p = ws + off;
    off += bytes;
    off = (off + 255) & ~(size_t)255;
    return p;
  };
  const size_t MB_ = (size_t)PB * PN;            // 4096 rows
  uint16_t* xb      = (uint16_t*)alloc(MB_ * PC * 2);
  uint16_t* peb     = (uint16_t*)alloc(MB_ * PC * 2);
  uint16_t* Wqkvb   = (uint16_t*)alloc((size_t)3 * PC * PC * 2);
  uint16_t* Wposb   = (uint16_t*)alloc((size_t)PC * PC * 2);
  uint16_t* Wprojb  = (uint16_t*)alloc((size_t)PC * PC * 2);
  uint16_t* qub     = (uint16_t*)alloc((size_t)PB * PH * PN * PHD * 2);
  uint16_t* qvb     = (uint16_t*)alloc((size_t)PB * PH * PN * PHD * 2);
  uint16_t* kbb     = (uint16_t*)alloc((size_t)PB * PH * PN * PHD * 2);
  uint16_t* vTb     = (uint16_t*)alloc((size_t)PB * PH * PHD * PN * 2);
  uint16_t* posb    = (uint16_t*)alloc((size_t)PB * PH * PN * PHD * 2);
  uint16_t* attn_in = (uint16_t*)alloc(MB_ * PC * 2);
  uint16_t* Sshift  = (uint16_t*)alloc((size_t)PB * PH * PN * PN * 2);  // fp16

  // 1) fp32 -> bf16 conversions
  {
    int nx = (int)(MB_ * PC);
    cvt_f32_bf16<<<(nx + 255) / 256, 256, 0, stream>>>(x,  xb,  nx);
    cvt_f32_bf16<<<(nx + 255) / 256, 256, 0, stream>>>(pe, peb, nx);
    int nw = 3 * PC * PC;
    cvt_f32_bf16<<<(nw + 255) / 256, 256, 0, stream>>>(Wqkv, Wqkvb, nw);
    int nw2 = PC * PC;
    cvt_f32_bf16<<<(nw2 + 255) / 256, 256, 0, stream>>>(Wpos,  Wposb,  nw2);
    cvt_f32_bf16<<<(nw2 + 255) / 256, 256, 0, stream>>>(Wproj, Wprojb, nw2);
  }
  // 2) QKV projection (fused u/v bias, head-major layouts, V transposed)
  {
    int waves = (int)((MB_ / 16) * ((3 * PC) / 64));     // 6144
    gemm_proj<<<waves / 4, 128, 0, stream>>>(xb, Wqkvb, (int)MB_, PC, 3 * PC, 0,
                                             ub, vb, nullptr,
                                             qub, qvb, kbb, vTb, nullptr, nullptr);
  }
  // 3) pos projection
  {
    int waves = (int)((MB_ / 16) * (PC / 64));           // 2048
    gemm_proj<<<waves / 4, 128, 0, stream>>>(peb, Wposb, (int)MB_, PC, PC, 1,
                                             nullptr, nullptr, nullptr,
                                             nullptr, nullptr, nullptr, nullptr,
                                             posb, nullptr);
  }
  // 4) pos-score GEMM with fused relative-shift scatter (fp16 S)
  {
    int waves = PB * PH * (PN / 16) * (PN / 64);         // 65536
    gemm_ps_shift<<<waves / 4, 128, 0, stream>>>(qvb, posb, Sshift);
  }
  // 5) flash attention (async double-buffered S tiles)
  {
    int waves = PB * PH * (PN / 16);                     // 2048
    flash_attn<<<waves / 4, 128, 0, stream>>>(qub, kbb, vTb, Sshift, attn_in);
  }
  // 6) output projection (+bias, fp32 result)
  {
    int waves = (int)((MB_ / 16) * (PC / 64));           // 2048
    gemm_proj<<<waves / 4, 128, 0, stream>>>(attn_in, Wprojb, (int)MB_, PC, PC, 2,
                                             nullptr, nullptr, bproj,
                                             nullptr, nullptr, nullptr, nullptr,
                                             nullptr, out);
  }
}
